// AttributeScatterMOE_14525579395178
// MI455X (gfx1250) — compile-verified, matrix operand order checked by reading, not run on hardware
//
#include <hip/hip_runtime.h>
#include <hip/hip_bf16.h>
#include <math.h>

// ---------------------------------------------------------------------------
// AttributeScatterMOE for MI455X (gfx1250, wave32, WMMA + async-to-LDS).
// - seq-len-1 cross attention folded: the two 77-GFLOP K/V projections become
//   two ~0.8-GFLOP streaming passes over visual_patchs (exact: softmax
//   shift-invariance kills bk; sum(att)=1 hoists Wv/bv).
// - expert+router GEMM (45 GFLOP) on v_wmma_f32_16x16x32_bf16. Both operands
//   are pre-swizzled into per-lane WMMA fragment order (bf16, tile-major), so
//   the GEMM hot loop is: global_load_async_to_lds_b128 (triple-buffered,
//   ASYNCcnt) -> ds_load_b128 fragments -> 16x WMMA. Epilogue fuses the MoE
//   gate combine (atomic) and router row-mean (shfl + atomic).
// ---------------------------------------------------------------------------

#define B_   256
#define N_   256
#define C_   768
#define A_   30
#define H_   8
#define E_   4
#define NC_  2
#define DH_  96
#define TOPK_ 21
#define KSTEPS 24          // 768 / 32

typedef __bf16 v16bf __attribute__((ext_vector_type(16)));
typedef float  v8f   __attribute__((ext_vector_type(8)));
typedef __attribute__((address_space(3))) __bf16 lds_bf16;

// workspace layout (float offsets)
#define WS_Q     0                         // [B,C]
#define WS_QK    196608                    // [B,H,C]
#define WS_O     1769472                   // [B,C]
#define WS_ATTR  1966080                   // [B,A,C] f32 (gate input)
#define WS_GW    7864320                   // [B,A,E]
#define WS_MOE   7895040                   // [B,A,C] (atomic accum)
#define WS_SCORE 13793280                  // [B*A]
#define WS_LOSS  13800960                  // [1]
#define WS_MU    13800964                  // [A,C]
#define WS_RSTD  13824004                  // [A,C]
#define WS_ATTRB 13847044                  // bf16 [60][24][8][32][16] swizzled A
#define WS_WB    16796164                  // bf16 [5][3][24][16][32][16] swizzled B
#define WS_END   19745284
#define ZERO_CNT (5898240 + 7680 + 1)      // MOE..LOSS contiguous

__device__ __forceinline__ float wred_sum(float v) {
#pragma unroll
  for (int off = 16; off; off >>= 1) v += __shfl_xor(v, off);
  return v;
}
__device__ __forceinline__ float wred_max(float v) {
#pragma unroll
  for (int off = 16; off; off >>= 1) v = fmaxf(v, __shfl_xor(v, off));
  return v;
}

// async 16B global -> LDS copy (ASYNCcnt-tracked, GV addressing)
__device__ __forceinline__ void async_cp16(unsigned lds, unsigned long long ga) {
  asm volatile("global_load_async_to_lds_b128 %0, %1, off"
               :: "v"(lds), "v"(ga) : "memory");
}

// ---- zero the atomic-accumulated regions (ws is poisoned 0xAA) -------------
__global__ void k_zero(float* __restrict__ p) {
  int i = blockIdx.x * 256 + threadIdx.x;
  if (i < ZERO_CNT) p[i] = 0.0f;
}

// ---- q = text_cls @ Wq + bq ------------------------------------------------
__global__ __launch_bounds__(256) void k_qproj(
    const float* __restrict__ text, const float* __restrict__ Wq,
    const float* __restrict__ bq, float* __restrict__ q_ws) {
  __shared__ float tl[C_];
  const int b = blockIdx.x, tid = threadIdx.x;
#pragma unroll
  for (int i = 0; i < 3; ++i) tl[tid + i * 256] = text[b * C_ + tid + i * 256];
  __syncthreads();
  const int j = blockIdx.y * 256 + tid;
  float acc = bq[j];
  for (int c = 0; c < C_; ++c) acc += tl[c] * Wq[c * C_ + j];
  q_ws[b * C_ + j] = acc;
}

// ---- qkvec[b,h,c] = scale * sum_d Wk[c, h*96+d] * q[b, h*96+d] -------------
__global__ __launch_bounds__(256) void k_qkvec(
    const float* __restrict__ q_ws, const float* __restrict__ Wk,
    float* __restrict__ qkvec) {
  __shared__ float qh[DH_];
  const int p = blockIdx.x;            // b*8 + h
  const int b = p >> 3, h = p & 7, tid = threadIdx.x;
  if (tid < DH_) qh[tid] = q_ws[b * C_ + h * DH_ + tid];
  __syncthreads();
  const int c = blockIdx.y * 256 + tid;
  const float* wrow = Wk + (size_t)c * C_ + h * DH_;
  float acc = 0.0f;
#pragma unroll 4
  for (int d = 0; d < DH_; ++d) acc += qh[d] * wrow[d];
  qkvec[(size_t)p * C_ + c] = acc * 0.10206207261596575f;  // 1/sqrt(96)
}

// ---- fused attention: scores -> softmax -> att-weighted x -> @Wv + bv ------
__global__ __launch_bounds__(256) void k_attn(
    const float* __restrict__ X, const float* __restrict__ qkvec,
    const float* __restrict__ Wv, const float* __restrict__ bv,
    float* __restrict__ o_ws) {
  __shared__ float xt[32][C_ + 1];
  __shared__ float s_l[H_][N_];
  __shared__ float xatt[H_][C_ + 4];
  const int b = blockIdx.x, tid = threadIdx.x;
  const int h = tid >> 5, lane = tid & 31;
  const float* xb = X + (size_t)b * N_ * C_;

  for (int t8 = 0; t8 < 8; ++t8) {
    __syncthreads();
#pragma unroll
    for (int i = 0; i < 24; ++i) {
      int idx = tid + i * 256;
      int row = idx / 192, c4 = (idx % 192) * 4;
      float4 v = *(const float4*)(xb + (size_t)(t8 * 32 + row) * C_ + c4);
      xt[row][c4] = v.x; xt[row][c4 + 1] = v.y;
      xt[row][c4 + 2] = v.z; xt[row][c4 + 3] = v.w;
    }
    __syncthreads();
    const float* qv = qkvec + ((size_t)b * H_ + h) * C_;
    float acc = 0.0f;
#pragma unroll 4
    for (int c = 0; c < C_; ++c) acc += xt[lane][c] * qv[c];
    s_l[h][t8 * 32 + lane] = acc;
  }
  __syncthreads();
  {
    float m = -3.4e38f, ev[8], s = 0.0f;
#pragma unroll
    for (int i = 0; i < 8; ++i) m = fmaxf(m, s_l[h][lane + i * 32]);
    m = wred_max(m);
#pragma unroll
    for (int i = 0; i < 8; ++i) { ev[i] = __expf(s_l[h][lane + i * 32] - m); s += ev[i]; }
    s = wred_sum(s);
    float inv = 1.0f / s;
#pragma unroll
    for (int i = 0; i < 8; ++i) s_l[h][lane + i * 32] = ev[i] * inv;
  }
  __syncthreads();
  for (int ch = 0; ch < 3; ++ch) {
    const int c = ch * 256 + tid;
    float a[H_];
#pragma unroll
    for (int hh = 0; hh < H_; ++hh) a[hh] = 0.0f;
    for (int n = 0; n < N_; ++n) {
      float xv = xb[(size_t)n * C_ + c];
#pragma unroll
      for (int hh = 0; hh < H_; ++hh) a[hh] += s_l[hh][n] * xv;
    }
#pragma unroll
    for (int hh = 0; hh < H_; ++hh) xatt[hh][c] = a[hh];
  }
  __syncthreads();
  for (int ch = 0; ch < 3; ++ch) {
    const int j = ch * 256 + tid;
    const int hh = j / DH_;
    float acc = bv[j];
#pragma unroll 4
    for (int c = 0; c < C_; ++c) acc += xatt[hh][c] * Wv[(size_t)c * C_ + j];
    o_ws[(size_t)b * C_ + j] = acc;
  }
}

// ---- moe_in = o @ Wo + bo ; attr_in = moe_in + prompt + visual_cls ---------
// Writes attr_in as f32 (gate) AND as fragment-swizzled bf16 (GEMM A operand).
__global__ __launch_bounds__(256) void k_moein(
    const float* __restrict__ o_ws, const float* __restrict__ Wo,
    const float* __restrict__ bo, const float* __restrict__ prompt,
    const float* __restrict__ vcls, float* __restrict__ attr_in,
    __bf16* __restrict__ attrB) {
  __shared__ float ol[C_];
  const int b = blockIdx.x, tid = threadIdx.x;
#pragma unroll
  for (int i = 0; i < 3; ++i) ol[tid + i * 256] = o_ws[(size_t)b * C_ + tid + i * 256];
  __syncthreads();
  for (int ch = 0; ch < 3; ++ch) {
    const int j = ch * 256 + tid;
    float mi = bo[j];
#pragma unroll 4
    for (int c = 0; c < C_; ++c) mi += ol[c] * Wo[(size_t)c * C_ + j];
    mi += vcls[(size_t)b * C_ + j];
    const int ks = j >> 5, kk = j & 31;
    const int l0 = (((kk >> 3) & 1) << 4);
    const int e0 = (kk & 7) | ((kk & 16) >> 1);
#pragma unroll
    for (int a = 0; a < A_; ++a) {
      float v = mi + prompt[a * C_ + j];
      const int row = b * A_ + a;
      attr_in[(size_t)row * C_ + j] = v;
      const int mb = row >> 7, tile = (row >> 4) & 7, m = row & 15;
      attrB[((((size_t)(mb * KSTEPS + ks) * 8 + tile) * 32) + m + l0) * 16 + e0] = (__bf16)v;
    }
  }
}

// ---- pre-swizzle {expert_W[0..3], router_W} into bf16 fragment order -------
// out[z][nb][ks][tileB][lane][e]; inverse of: lane=n%16+16*K[3], e=K[2:0]|K[4]<<3
__global__ __launch_bounds__(256) void k_wswz(
    const float* __restrict__ expert_W, const float* __restrict__ router_W,
    __bf16* __restrict__ wb) {
  const int o = blockIdx.x * 256 + threadIdx.x;   // < 5*3*24*8192
  const int e = o & 15;
  const int lane = (o >> 4) & 31;
  const int tb = (o >> 9) & 15;
  const int r = o >> 13;                          // z*72 + nb*24 + ks
  const int ks = r % 24, nb = (r / 24) % 3, z = r / 72;
  const int kk = (e & 7) | (((lane >> 4) & 1) << 3) | ((e & 8) << 1);
  const int k = ks * 32 + kk;
  const int n = nb * 256 + tb * 16 + (lane & 15);
  const float* src = (z < 4) ? (expert_W + (size_t)z * C_ * C_) : router_W;
  wb[o] = (__bf16)src[(size_t)k * C_ + n];
}

// ---- gate: top-3 of 4 experts, softmax over selected -----------------------
__global__ __launch_bounds__(256) void k_gate(
    const float* __restrict__ attr_in, const float* __restrict__ gate_W,
    const float* __restrict__ gate_b, float* __restrict__ gw) {
  const int p = blockIdx.x * 8 + (threadIdx.x >> 5);
  const int lane = threadIdx.x & 31;
  float pe[E_] = {0.f, 0.f, 0.f, 0.f};
  for (int c = lane; c < C_; c += 32) {
    float x = attr_in[(size_t)p * C_ + c];
#pragma unroll
    for (int e = 0; e < E_; ++e) pe[e] += x * gate_W[c * E_ + e];
  }
#pragma unroll
  for (int e = 0; e < E_; ++e) pe[e] = wred_sum(pe[e]);
  if (lane == 0) {
    float gl[E_];
#pragma unroll
    for (int e = 0; e < E_; ++e) gl[e] = pe[e] + gate_b[e];
    int excl = 0;
#pragma unroll
    for (int e = 1; e < E_; ++e) if (gl[e] <= gl[excl]) excl = e;
    float m = -3.4e38f;
#pragma unroll
    for (int e = 0; e < E_; ++e) if (e != excl) m = fmaxf(m, gl[e]);
    float s = 0.f, w[E_];
#pragma unroll
    for (int e = 0; e < E_; ++e) { w[e] = (e != excl) ? __expf(gl[e] - m) : 0.f; s += w[e]; }
    float inv = 1.f / s;
#pragma unroll
    for (int e = 0; e < E_; ++e) gw[p * E_ + e] = w[e] * inv;
  }
}

// ---- WMMA GEMM: swizzled-bf16 A[7680x768] x {expert_W[e], router_W} --------
// grid (3, 60, 5). Triple-buffered async global->LDS staging; hot loop is
// global_load_async_to_lds_b128 -> s_wait_asynccnt -> barrier -> ds_load_b128
// fragments -> 16x v_wmma_f32_16x16x32_bf16. One barrier per K-step.
__device__ __forceinline__ void stage_tiles(unsigned ldsA0, unsigned ldsB0, int buf,
                                            const __bf16* aSrc, const __bf16* bSrc,
                                            int ks, int tid) {
  const unsigned la = ldsA0 + buf * 8192u + tid * 16u;
  const unsigned lb = ldsB0 + buf * 16384u + tid * 16u;
  const unsigned long long ga = (unsigned long long)(aSrc + (size_t)ks * 4096 + tid * 8);
  const unsigned long long gb = (unsigned long long)(bSrc + (size_t)ks * 8192 + tid * 8);
  async_cp16(la, ga);
  async_cp16(la + 4096u, ga + 4096u);
  async_cp16(lb, gb);
  async_cp16(lb + 4096u, gb + 4096u);
  async_cp16(lb + 8192u, gb + 8192u);
  async_cp16(lb + 12288u, gb + 12288u);
}

__global__ __launch_bounds__(256) void k_gemm_moe(
    const __bf16* __restrict__ attrB, const __bf16* __restrict__ wswz,
    const float* __restrict__ expert_b, const float* __restrict__ gw,
    float* __restrict__ moe_out, float* __restrict__ score_raw) {
  __shared__ __align__(16) __bf16 ldsA[3][4096];   // 3 x 8KB  (128x32 tile)
  __shared__ __align__(16) __bf16 ldsB[3][8192];   // 3 x 16KB (32x256 tile)
  const int tid = threadIdx.x, wave = tid >> 5, lane = tid & 31;
  const int z = blockIdx.z, mb = blockIdx.y, nb = blockIdx.x;
  const int waveM = (wave >> 2) * 64, waveN = (wave & 3) * 64;

  const unsigned ldsA0 = (unsigned)(unsigned long long)(lds_bf16*)&ldsA[0][0];
  const unsigned ldsB0 = (unsigned)(unsigned long long)(lds_bf16*)&ldsB[0][0];
  const __bf16* aSrc = attrB + (size_t)mb * KSTEPS * 4096;
  const __bf16* bSrc = wswz + (size_t)(z * 3 + nb) * KSTEPS * 8192;

  v8f acc[4][4] = {};

  stage_tiles(ldsA0, ldsB0, 0, aSrc, bSrc, 0, tid);
  for (int ks = 0; ks < KSTEPS; ++ks) {
    const int buf = ks % 3;
    if (ks + 1 < KSTEPS) {
      stage_tiles(ldsA0, ldsB0, (ks + 1) % 3, aSrc, bSrc, ks + 1, tid);
      asm volatile("s_wait_asynccnt 0x6" ::: "memory");   // prev stage done
    } else {
      asm volatile("s_wait_asynccnt 0x0" ::: "memory");
    }
    __syncthreads();
    const __bf16* lA = &ldsA[buf][0];
    const __bf16* lB = &ldsB[buf][0];
    v16bf afr[4], bfr[4];
#pragma unroll
    for (int i = 0; i < 4; ++i)
      afr[i] = *(const v16bf*)&lA[((waveM >> 4) + i) * 512 + lane * 16];
#pragma unroll
    for (int j = 0; j < 4; ++j)
      bfr[j] = *(const v16bf*)&lB[((waveN >> 4) + j) * 512 + lane * 16];
#pragma unroll
    for (int i = 0; i < 4; ++i)
#pragma unroll
      for (int j = 0; j < 4; ++j)
        acc[i][j] = __builtin_amdgcn_wmma_f32_16x16x32_bf16(
            false, afr[i], false, bfr[j], (short)0, acc[i][j], false, false);
  }

  const int mh = (lane >> 4) * 8, nl = lane & 15;
  const int m0 = mb * 128, n0 = nb * 256;
  if (z < 4) {
#pragma unroll
    for (int i = 0; i < 4; ++i)
#pragma unroll
      for (int r = 0; r < 8; ++r) {
        int row = m0 + waveM + i * 16 + mh + r;
        float g = gw[row * E_ + z];
#pragma unroll
        for (int j = 0; j < 4; ++j) {
          int col = n0 + waveN + j * 16 + nl;
          float val = acc[i][j][r] + expert_b[z * C_ + col];
          atomicAdd(&moe_out[(size_t)row * C_ + col], g * val);
        }
      }
  } else {
#pragma unroll
    for (int i = 0; i < 4; ++i)
#pragma unroll
      for (int r = 0; r < 8; ++r) {
        float rs = acc[i][0][r] + acc[i][1][r] + acc[i][2][r] + acc[i][3][r];
#pragma unroll
        for (int off = 1; off < 16; off <<= 1) rs += __shfl_xor(rs, off);
        if (nl == 0) {
          int row = m0 + waveM + i * 16 + mh + r;
          atomicAdd(&score_raw[row], rs);
        }
      }
  }
}

// ---- batchnorm stats over B per (a,c) --------------------------------------
__global__ __launch_bounds__(256) void k_bnstats(
    const float* __restrict__ moe_out, float* __restrict__ mu_ws,
    float* __restrict__ rstd_ws) {
  const int idx = blockIdx.x * 256 + threadIdx.x;
  float s = 0.f, ss = 0.f;
  for (int b = 0; b < B_; ++b) {
    float v = moe_out[(size_t)b * (A_ * C_) + idx];
    s += v; ss += v * v;
  }
  float mu = s * (1.f / B_);
  float var = ss * (1.f / B_) - mu * mu;
  mu_ws[idx] = mu;
  rstd_ws[idx] = rsqrtf(var + 1e-5f);
}

// ---- BN-affine + classifier + cross-entropy (NC=2) -------------------------
__global__ __launch_bounds__(256) void k_loss(
    const float* __restrict__ moe_out, const float* __restrict__ mu_ws,
    const float* __restrict__ rstd_ws, const float* __restrict__ bn_w,
    const float* __restrict__ bn_b, const float* __restrict__ cls_W,
    const int* __restrict__ labels, float* __restrict__ loss_ws) {
  const int p = blockIdx.x * 8 + (threadIdx.x >> 5);
  const int lane = threadIdx.x & 31;
  const int a = p % A_;
  float z0 = 0.f, z1 = 0.f;
  for (int c = lane; c < C_; c += 32) {
    int ai = a * C_ + c;
    float fb = (moe_out[(size_t)p * C_ + c] - mu_ws[ai]) * rstd_ws[ai] * bn_w[ai] + bn_b[ai];
    z0 += fb * cls_W[(size_t)(a * NC_) * C_ + c];
    z1 += fb * cls_W[(size_t)(a * NC_ + 1) * C_ + c];
  }
  z0 = wred_sum(z0);
  z1 = wred_sum(z1);
  if (lane == 0) {
    float m = fmaxf(z0, z1);
    float lse = m + __logf(__expf(z0 - m) + __expf(z1 - m));
    float zt = labels[p] ? z1 : z0;
    atomicAdd(loss_ws, (lse - zt) * (1.f / B_));
  }
}

// ---- top-21 attrs by router score, softmax combine -> enhanced_feat --------
__global__ __launch_bounds__(256) void k_topk(
    const float* __restrict__ moe_out, const float* __restrict__ score_raw,
    const float* __restrict__ loss_ws, float* __restrict__ out) {
  __shared__ float sc[A_], w[A_];
  __shared__ int fl[A_];
  const int b = blockIdx.x, tid = threadIdx.x;
  if (tid < A_) sc[tid] = score_raw[b * A_ + tid] * (1.f / C_);
  __syncthreads();
  if (tid < A_) {
    float v = sc[tid];
    int rank = 0;
    for (int j = 0; j < A_; ++j) {
      float u = sc[j];
      rank += (u > v) || (u == v && j < tid);
    }
    fl[tid] = (rank < TOPK_);
  }
  __syncthreads();
  if (tid == 0) {
    float m = -3.4e38f;
    for (int j = 0; j < A_; ++j) if (fl[j]) m = fmaxf(m, sc[j]);
    float s = 0.f;
    for (int j = 0; j < A_; ++j) { w[j] = fl[j] ? __expf(sc[j] - m) : 0.f; s += w[j]; }
    float inv = 1.f / s;
    for (int j = 0; j < A_; ++j) w[j] *= inv;
  }
  __syncthreads();
  for (int ch = 0; ch < 3; ++ch) {
    const int c = ch * 256 + tid;
    float acc = 0.f;
#pragma unroll 5
    for (int a = 0; a < A_; ++a)
      acc += w[a] * moe_out[((size_t)b * A_ + a) * C_ + c];
    out[(size_t)b * C_ + c] = acc;
  }
  if (b == 0 && tid == 0) out[B_ * C_] = loss_ws[0];
}

// ---------------------------------------------------------------------------
extern "C" void kernel_launch(void* const* d_in, const int* in_sizes, int n_in,
                              void* d_out, int out_size, void* d_ws, size_t ws_size,
                              hipStream_t stream) {
  (void)in_sizes; (void)n_in; (void)out_size; (void)ws_size;
  const float* text   = (const float*)d_in[0];
  const float* vcls   = (const float*)d_in[1];
  const float* X      = (const float*)d_in[2];
  const float* prompt = (const float*)d_in[3];
  const float* Wq = (const float*)d_in[4];  const float* bq = (const float*)d_in[5];
  const float* Wk = (const float*)d_in[6];  /* bk: softmax-invariant, unused */
  const float* Wv = (const float*)d_in[8];  const float* bv = (const float*)d_in[9];
  const float* Wo = (const float*)d_in[10]; const float* bo = (const float*)d_in[11];
  const float* gate_W = (const float*)d_in[12]; const float* gate_b = (const float*)d_in[13];
  const float* expert_W = (const float*)d_in[14]; const float* expert_b = (const float*)d_in[15];
  const float* router_W = (const float*)d_in[16]; /* router_b: shift-invariant, unused */
  const float* bn_w = (const float*)d_in[18]; const float* bn_b = (const float*)d_in[19];
  const float* cls_W = (const float*)d_in[20];
  const int*   labels = (const int*)d_in[21];
  float* out = (float*)d_out;
  float* ws  = (float*)d_ws;
  __bf16* attrB = (__bf16*)(ws + WS_ATTRB);
  __bf16* wswz  = (__bf16*)(ws + WS_WB);

  k_zero<<<(ZERO_CNT + 255) / 256, 256, 0, stream>>>(ws + WS_MOE);
  k_wswz<<<(5 * 3 * KSTEPS * 8192) / 256, 256, 0, stream>>>(expert_W, router_W, wswz);
  k_qproj<<<dim3(B_, 3), 256, 0, stream>>>(text, Wq, bq, ws + WS_Q);
  k_qkvec<<<dim3(B_ * H_, 3), 256, 0, stream>>>(ws + WS_Q, Wk, ws + WS_QK);
  k_attn<<<B_, 256, 0, stream>>>(X, ws + WS_QK, Wv, bv, ws + WS_O);
  k_moein<<<B_, 256, 0, stream>>>(ws + WS_O, Wo, bo, prompt, vcls,
                                  ws + WS_ATTR, attrB);
  k_gate<<<(B_ * A_) / 8, 256, 0, stream>>>(ws + WS_ATTR, gate_W, gate_b, ws + WS_GW);
  k_gemm_moe<<<dim3(3, 60, 5), 256, 0, stream>>>(attrB, wswz, expert_b,
                                                 ws + WS_GW, ws + WS_MOE, ws + WS_SCORE);
  k_bnstats<<<(A_ * C_) / 256, 256, 0, stream>>>(ws + WS_MOE, ws + WS_MU, ws + WS_RSTD);
  k_loss<<<(B_ * A_) / 8, 256, 0, stream>>>(ws + WS_MOE, ws + WS_MU, ws + WS_RSTD,
                                            bn_w, bn_b, cls_W, labels, ws + WS_LOSS);
  k_topk<<<B_, 256, 0, stream>>>(ws + WS_MOE, ws + WS_SCORE, ws + WS_LOSS, out);
}
